// Attention_19739669692550
// MI455X (gfx1250) — compile-verified, matrix operand order checked by reading, not run on hardware
//
#include <hip/hip_runtime.h>
#include <hip/hip_bf16.h>

// ---------------- problem constants (match reference) ----------------
#define B_  2
#define S_  2048
#define D_  2048
#define H_  32
#define KVH_ 8
#define HD_ 64
#define NREP_ 4          // H / KVH

typedef __attribute__((ext_vector_type(16))) __bf16 v16bf;
typedef __attribute__((ext_vector_type(8)))  float  v8f;
typedef int v4i_ld __attribute__((vector_size(16)));           // matches builtin param
typedef __attribute__((address_space(3))) v4i_ld* lds_v4i_ptr;

union Frag { v16bf bf; unsigned int u[8]; };

#if defined(__has_builtin)
#if __has_builtin(__builtin_amdgcn_global_load_async_to_lds_b128)
#define HAS_ASYNC_LDS 1
#endif
#endif

__device__ __forceinline__ v8f zero8() {
    v8f z = {0.f,0.f,0.f,0.f,0.f,0.f,0.f,0.f};
    return z;
}

__device__ __forceinline__ v8f wmma_bf16(const Frag& a, const Frag& b, v8f c) {
    // (neg_a, A, neg_b, B, c_mod, C, reuse_a, reuse_b)
    return __builtin_amdgcn_wmma_f32_16x16x32_bf16(false, a.bf, false, b.bf,
                                                   (short)0, c, false, false);
}

__device__ __forceinline__ unsigned short f2bf(float f) {
    unsigned int u = __float_as_uint(f);
    unsigned int r = (u + 0x7FFFu + ((u >> 16) & 1u)) >> 16;   // RNE
    return (unsigned short)r;
}

// K-pair base within a 16-bit A/B fragment, per CDNA5 ISA 16-bit A 16x32 layout:
// lanes 0-15:  VGPR0-3 -> K 0..7,  VGPR4-7 -> K 16..23
// lanes 16-31: VGPR0-3 -> K 8..15, VGPR4-7 -> K 24..31
__device__ __forceinline__ int kpair(int i, int hf) {
    return ((i < 4) ? 0 : 16) + hf * 8 + (i & 3) * 2;
}

// ---------------- elementwise converts ----------------
__global__ void cvt_f32_bf16(const float* __restrict__ in,
                             unsigned short* __restrict__ out, int n) {
    int i = blockIdx.x * blockDim.x + threadIdx.x;
    if (i < n) out[i] = f2bf(in[i]);
}

// proj: [B*S, nh*HD] f32  ->  out: [B, nh, S, HD] bf16, with RoPE on (even,odd) pairs
__global__ void rope_pack(const float* __restrict__ proj,
                          const float* __restrict__ cosT,
                          const float* __restrict__ sinT,
                          unsigned short* __restrict__ out, int nh) {
    int i = blockIdx.x * blockDim.x + threadIdx.x;
    int total = B_ * S_ * nh * (HD_ / 2);
    if (i >= total) return;
    int j = i % (HD_ / 2); int t = i / (HD_ / 2);
    int h = t % nh; t /= nh;
    int s = t % S_; int b = t / S_;
    const float* row = proj + (size_t)(b * S_ + s) * (nh * HD_) + h * HD_;
    float a  = row[2 * j];
    float bb = row[2 * j + 1];
    float c  = cosT[s * (HD_ / 2) + j];
    float sn = sinT[s * (HD_ / 2) + j];
    size_t o = ((size_t)(b * nh + h) * S_ + s) * HD_ + 2 * j;
    out[o]     = f2bf(a * c - bb * sn);
    out[o + 1] = f2bf(a * sn + bb * c);
}

// vf: [B*S, KVH*HD] f32 -> Vb: [B, KVH, S, HD] bf16
__global__ void v_pack(const float* __restrict__ vf,
                       unsigned short* __restrict__ Vb) {
    int i = blockIdx.x * blockDim.x + threadIdx.x;
    int total = B_ * S_ * KVH_ * HD_;
    if (i >= total) return;
    int hd = i % HD_; int t = i / HD_;
    int kv = t % KVH_; t /= KVH_;
    int s = t % S_; int b = t / S_;
    float v = vf[(size_t)(b * S_ + s) * (KVH_ * HD_) + kv * HD_ + hd];
    Vb[((size_t)(b * KVH_ + kv) * S_ + s) * HD_ + hd] = f2bf(v);
}

// ---------------- bf16 WMMA GEMM: C[M,N] = A[M,K] x B[K,N] ----------------
// A,B bf16 row-major, C f32. Block: 256 thr = 8 waves; block tile 128x128,
// each wave owns a 32x64 sub-tile (2 A-frags x 4 B-frags = 8 WMMAs / K-step).
// LDS rows are 80B (16B-aligned for async b128; 20-dword advance spreads banks).
#define ASTR 40   // ushorts per LDS row (80 bytes)
__global__ __launch_bounds__(256)
void gemm_bf16_wmma(const unsigned short* __restrict__ A,
                    const unsigned short* __restrict__ Bm,
                    float* __restrict__ C, int M, int N, int K) {
    __shared__ unsigned short As[128 * ASTR];
    __shared__ unsigned short Bs[128 * ASTR];   // holds B^T tile: Bs[n][k]
    int t = threadIdx.x;
    int lane = t & 31, w = t >> 5;
    int hf = lane >> 4, r = lane & 15;
    int m0 = blockIdx.y * 128, n0 = blockIdx.x * 128;
    int wm = (w >> 1) * 32, wn = (w & 1) * 64;

    v8f acc[2][4];
    #pragma unroll
    for (int i = 0; i < 2; i++)
        #pragma unroll
        for (int j = 0; j < 4; j++) acc[i][j] = zero8();

    int arow = t >> 1, ah = t & 1;   // A loader: 128 rows x 2 half-chunks(16)
    int brow = t >> 3, bq = t & 7;   // B loader: 32 rows x 8 col-chunks(16)

    for (int kk = 0; kk < K; kk += 32) {
        const unsigned short* agp = A + (size_t)(m0 + arow) * K + kk + ah * 16;
        const unsigned short* bgp = Bm + (size_t)(kk + brow) * N + n0 + bq * 16;
#if !defined(HAS_ASYNC_LDS)
        uint4 av0 = ((const uint4*)agp)[0];
        uint4 av1 = ((const uint4*)agp)[1];
#endif
        uint4 bv0 = ((const uint4*)bgp)[0];
        uint4 bv1 = ((const uint4*)bgp)[1];
        if (kk + 32 < K) {  // global_prefetch_b8 of next tiles
            __builtin_prefetch(agp + 32, 0, 0);
            __builtin_prefetch(Bm + (size_t)(kk + 32 + brow) * N + n0 + bq * 16, 0, 0);
        }
        __syncthreads();   // previous iteration's LDS reads are done

        unsigned short* adst = &As[arow * ASTR + ah * 16];
#if defined(HAS_ASYNC_LDS)
        // CDNA5 async copy: global -> LDS with no VGPR round-trip (ASYNCcnt).
        __builtin_amdgcn_global_load_async_to_lds_b128(
            (v4i_ld*)agp, (lds_v4i_ptr)adst, 0, 0);
        __builtin_amdgcn_global_load_async_to_lds_b128(
            (v4i_ld*)(agp + 8), (lds_v4i_ptr)(adst + 8), 0, 0);
#else
        unsigned int* asd = (unsigned int*)adst;
        asd[0] = av0.x; asd[1] = av0.y; asd[2] = av0.z; asd[3] = av0.w;
        asd[4] = av1.x; asd[5] = av1.y; asd[6] = av1.z; asd[7] = av1.w;
#endif
        unsigned int bu[8] = {bv0.x, bv0.y, bv0.z, bv0.w, bv1.x, bv1.y, bv1.z, bv1.w};
        #pragma unroll
        for (int j = 0; j < 8; j++) {   // transpose into Bs
            Bs[(bq * 16 + 2 * j    ) * ASTR + brow] = (unsigned short)(bu[j] & 0xffffu);
            Bs[(bq * 16 + 2 * j + 1) * ASTR + brow] = (unsigned short)(bu[j] >> 16);
        }
#if defined(HAS_ASYNC_LDS)
#if __has_builtin(__builtin_amdgcn_s_wait_asynccnt)
        __builtin_amdgcn_s_wait_asynccnt(0);
#else
        asm volatile("s_wait_asynccnt 0" ::: "memory");
#endif
#endif
        __syncthreads();

        Frag a0, a1, b[4];
        #pragma unroll
        for (int i = 0; i < 8; i++) {
            int kp = kpair(i, hf);
            a0.u[i] = *(const unsigned int*)&As[(wm +      r) * ASTR + kp];
            a1.u[i] = *(const unsigned int*)&As[(wm + 16 + r) * ASTR + kp];
            #pragma unroll
            for (int j = 0; j < 4; j++)
                b[j].u[i] = *(const unsigned int*)&Bs[(wn + j * 16 + r) * ASTR + kp];
        }
        #pragma unroll
        for (int j = 0; j < 4; j++) {
            acc[0][j] = wmma_bf16(a0, b[j], acc[0][j]);
            acc[1][j] = wmma_bf16(a1, b[j], acc[1][j]);
        }
    }
    // C layout: VGPR rr -> row rr (lanes 0-15) / rr+8 (lanes 16-31), col = lane&15
    #pragma unroll
    for (int mi = 0; mi < 2; mi++)
        #pragma unroll
        for (int ni = 0; ni < 4; ni++)
            #pragma unroll
            for (int rr = 0; rr < 8; rr++) {
                int row = m0 + wm + mi * 16 + rr + hf * 8;
                int col = n0 + wn + ni * 16 + r;
                C[(size_t)row * N + col] = acc[mi][ni][rr];
            }
}

// ---------------- flash attention ----------------
// 4 waves / block, each wave owns one (b, h, 16-query tile); waves in a block
// cover adjacent q-tiles of the SAME head so K/V streams hit in WGP$.
// Q,K,V bf16 head-major [B,{H|KVH},S,HD]; Out bf16 [B*S, H*HD].
#define TPAD 34
__global__ __launch_bounds__(128)
void flash_attn_wmma(const unsigned short* __restrict__ Q,
                     const unsigned short* __restrict__ Kc,
                     const unsigned short* __restrict__ Vc,
                     unsigned short* __restrict__ Out) {
    __shared__ unsigned short Pl[4][16 * TPAD];
    int w = threadIdx.x >> 5;
    int lane = threadIdx.x & 31;
    int hf = lane >> 4, r = lane & 15;
    int qt = blockIdx.x * 4 + w, h = blockIdx.y, b = blockIdx.z;
    int kvh = h / NREP_;
    int qbase = qt * 16;
    const unsigned short* Qp = Q  + ((size_t)(b * H_   + h  ) * S_ + qbase) * HD_;
    const unsigned short* Kp = Kc +  (size_t)(b * KVH_ + kvh) * S_ * HD_;
    const unsigned short* Vp = Vc +  (size_t)(b * KVH_ + kvh) * S_ * HD_;

    // Q fragments (16 x 64 = two 16x32 A-frags) stay in registers
    Frag qa[2];
    #pragma unroll
    for (int t2 = 0; t2 < 2; t2++)
        #pragma unroll
        for (int i = 0; i < 8; i++)
            qa[t2].u[i] = *(const unsigned int*)(Qp + (size_t)r * HD_ + t2 * 32 + kpair(i, hf));

    float m[8], l[8];
    v8f acc[4];
    #pragma unroll
    for (int i = 0; i < 8; i++) { m[i] = -1e30f; l[i] = 0.f; }
    #pragma unroll
    for (int n = 0; n < 4; n++) acc[n] = zero8();

    int nch = (qbase + 15) / 32 + 1;     // causal limit: key chunks of 32
    for (int c = 0; c < nch; c++) {
        int kb = c * 32;
        // ---- scores: Q(16x64) @ K^T(64x32) -> two 16x16 C tiles ----
        v8f sc[2];
        #pragma unroll
        for (int nt = 0; nt < 2; nt++) {
            Frag kf0, kf1;   // B-frag == A-frag of (K^T)^T = rows of K: direct loads
            const unsigned short* krow = Kp + (size_t)(kb + nt * 16 + r) * HD_;
            #pragma unroll
            for (int i = 0; i < 8; i++) {
                int kp = kpair(i, hf);
                kf0.u[i] = *(const unsigned int*)(krow + kp);
                kf1.u[i] = *(const unsigned int*)(krow + 32 + kp);
            }
            v8f z = zero8();
            z = wmma_bf16(qa[0], kf0, z);
            z = wmma_bf16(qa[1], kf1, z);
            sc[nt] = z;
        }
        // ---- scale + causal mask + online softmax (rows across 16 lanes) ----
        float p0[8], p1[8];
        #pragma unroll
        for (int rr = 0; rr < 8; rr++) {
            int qrow = qbase + rr + hf * 8;
            float s0 = sc[0][rr] * 0.125f + ((kb + r)      <= qrow ? 0.f : -1e9f);
            float s1 = sc[1][rr] * 0.125f + ((kb + 16 + r) <= qrow ? 0.f : -1e9f);
            float mx = fmaxf(s0, s1);
            #pragma unroll
            for (int d = 1; d < 16; d <<= 1) mx = fmaxf(mx, __shfl_xor(mx, d, 32));
            float mn = fmaxf(m[rr], mx);
            p0[rr] = __expf(s0 - mn);
            p1[rr] = __expf(s1 - mn);
            float sum = p0[rr] + p1[rr];
            #pragma unroll
            for (int d = 1; d < 16; d <<= 1) sum += __shfl_xor(sum, d, 32);
            float alpha = __expf(m[rr] - mn);
            l[rr] = l[rr] * alpha + sum;
            m[rr] = mn;
            #pragma unroll
            for (int n = 0; n < 4; n++) acc[n][rr] *= alpha;
        }
        // ---- P: C-layout -> A-layout via per-wave LDS bounce ----
        #pragma unroll
        for (int rr = 0; rr < 8; rr++) {
            Pl[w][(rr + hf * 8) * TPAD + r]      = f2bf(p0[rr]);
            Pl[w][(rr + hf * 8) * TPAD + 16 + r] = f2bf(p1[rr]);
        }
        asm volatile("s_wait_dscnt 0" ::: "memory");   // same-wave RAW fence
        Frag pa;
        #pragma unroll
        for (int i = 0; i < 8; i++)
            pa.u[i] = *(const unsigned int*)&Pl[w][r * TPAD + kpair(i, hf)];
        // ---- O += P(16x32) @ V(32x64): four 16x16 N-tiles ----
        #pragma unroll
        for (int n = 0; n < 4; n++) {
            Frag vf;   // B-frag: col = hd (lane&15), K = key index
            #pragma unroll
            for (int i = 0; i < 8; i++) {
                int kp = kpair(i, hf);
                unsigned int lo = Vp[(size_t)(kb + kp)     * HD_ + n * 16 + r];
                unsigned int hi = Vp[(size_t)(kb + kp + 1) * HD_ + n * 16 + r];
                vf.u[i] = lo | (hi << 16);
            }
            acc[n] = wmma_bf16(pa, vf, acc[n]);
        }
    }
    // ---- finalize: O /= l, store bf16 [B*S, H*HD] for output projection ----
    #pragma unroll
    for (int rr = 0; rr < 8; rr++) {
        float inv = 1.0f / l[rr];
        size_t row = (size_t)b * S_ + qbase + rr + hf * 8;
        #pragma unroll
        for (int n = 0; n < 4; n++)
            Out[row * (H_ * HD_) + h * HD_ + n * 16 + r] = f2bf(acc[n][rr] * inv);
    }
}

// ---------------- host-side orchestration ----------------
extern "C" void kernel_launch(void* const* d_in, const int* in_sizes, int n_in,
                              void* d_out, int out_size, void* d_ws, size_t ws_size,
                              hipStream_t stream) {
    const float* x    = (const float*)d_in[0];
    const float* wq   = (const float*)d_in[1];
    const float* wk   = (const float*)d_in[2];
    const float* wv   = (const float*)d_in[3];
    const float* wo   = (const float*)d_in[4];
    const float* cosT = (const float*)d_in[5];
    const float* sinT = (const float*)d_in[6];
    // d_in[7] (mask) unused: causal mask computed arithmetically.

    const size_t MB = 1u << 20;
    char* ws = (char*)d_ws;
    unsigned short* xb   = (unsigned short*)(ws + 0 * MB);     // 16 MB
    unsigned short* wqb  = (unsigned short*)(ws + 16 * MB);    //  8 MB
    unsigned short* wkb  = (unsigned short*)(ws + 24 * MB);    //  2 MB
    unsigned short* wvb  = (unsigned short*)(ws + 26 * MB);    //  2 MB
    unsigned short* wob  = (unsigned short*)(ws + 28 * MB);    //  8 MB
    float*          qf   = (float*)        (ws + 36 * MB);     // 32 MB
    float*          kf   = (float*)        (ws + 68 * MB);     //  8 MB
    float*          vf   = (float*)        (ws + 76 * MB);     //  8 MB
    unsigned short* Qb   = (unsigned short*)(ws + 84 * MB);    // 16 MB
    unsigned short* Kb   = (unsigned short*)(ws + 100 * MB);   //  4 MB
    unsigned short* Vb   = (unsigned short*)(ws + 104 * MB);   //  4 MB
    unsigned short* attb = (unsigned short*)(ws + 36 * MB);    // aliases qf (dead)

    const int nx  = B_ * S_ * D_;          // 8M
    const int nwq = D_ * H_ * HD_;         // 4M
    const int nwk = D_ * KVH_ * HD_;       // 1M
    const int BS  = B_ * S_;               // 4096

    cvt_f32_bf16<<<(nx  + 255) / 256, 256, 0, stream>>>(x,  xb,  nx);
    cvt_f32_bf16<<<(nwq + 255) / 256, 256, 0, stream>>>(wq, wqb, nwq);
    cvt_f32_bf16<<<(nwk + 255) / 256, 256, 0, stream>>>(wk, wkb, nwk);
    cvt_f32_bf16<<<(nwk + 255) / 256, 256, 0, stream>>>(wv, wvb, nwk);
    cvt_f32_bf16<<<(nwq + 255) / 256, 256, 0, stream>>>(wo, wob, nwq);

    // projections (block tile 128x128)
    gemm_bf16_wmma<<<dim3(D_ / 128, BS / 128), 256, 0, stream>>>(xb, wqb, qf, BS, H_ * HD_,   D_);
    gemm_bf16_wmma<<<dim3((KVH_ * HD_) / 128, BS / 128), 256, 0, stream>>>(xb, wkb, kf, BS, KVH_ * HD_, D_);
    gemm_bf16_wmma<<<dim3((KVH_ * HD_) / 128, BS / 128), 256, 0, stream>>>(xb, wvb, vf, BS, KVH_ * HD_, D_);

    // RoPE + head-major bf16 packing
    int nq2 = B_ * S_ * H_   * (HD_ / 2);
    int nk2 = B_ * S_ * KVH_ * (HD_ / 2);
    int nv  = B_ * S_ * KVH_ * HD_;
    rope_pack<<<(nq2 + 255) / 256, 256, 0, stream>>>(qf, cosT, sinT, Qb, H_);
    rope_pack<<<(nk2 + 255) / 256, 256, 0, stream>>>(kf, cosT, sinT, Kb, KVH_);
    v_pack<<<(nv + 255) / 256, 256, 0, stream>>>(vf, Vb);

    // attention (after this, qf region is dead -> attb aliases it)
    flash_attn_wmma<<<dim3(S_ / 64, H_, B_), 128, 0, stream>>>(Qb, Kb, Vb, attb);

    // output projection -> f32 d_out
    gemm_bf16_wmma<<<dim3(D_ / 128, BS / 128), 256, 0, stream>>>(attb, wob, (float*)d_out,
                                                                 BS, D_, D_);
}
